// TransformerMixerBlock_87170656240121
// MI455X (gfx1250) — compile-verified, matrix operand order checked by reading, not run on hardware
//
#include <hip/hip_runtime.h>
#include <hip/hip_bf16.h>
#include <math.h>

// Problem constants (from reference)
#define B_  2
#define S_  2048
#define D_  1024
#define H_  16
#define HD_ 64
#define M_  (B_ * S_)      // 4096 total rows (b,s flattened)
#define QKVN_ (3 * D_)     // 3072

typedef __bf16 bf16;
typedef __attribute__((ext_vector_type(16))) __bf16 v16bf;
typedef __attribute__((ext_vector_type(8)))  float  v8f;
typedef __attribute__((ext_vector_type(4)))  unsigned int u32x4;
typedef __attribute__((ext_vector_type(4)))  int i32x4;

#define GLOBAL_AS __attribute__((address_space(1)))
#define LDS_AS    __attribute__((address_space(3)))

#if defined(__has_builtin)
#if __has_builtin(__builtin_amdgcn_global_load_async_to_lds_b128)
#define HAVE_ASYNC_LDS 1
#else
#define HAVE_ASYNC_LDS 0
#endif
#else
#define HAVE_ASYNC_LDS 0
#endif

__device__ __forceinline__ unsigned short f2bf(float f) {
  unsigned int u = __float_as_uint(f);
  unsigned int r = 0x7FFFu + ((u >> 16) & 1u);   // round-to-nearest-even
  return (unsigned short)((u + r) >> 16);
}
__device__ __forceinline__ float bf2f(unsigned short u) {
  return __uint_as_float(((unsigned int)u) << 16);
}

// One 16-bf16 WMMA operand fragment as two 16-byte vector loads,
// bit-punned directly into the operand register tuple (no per-element moves).
// p0 supplies K[khalf*8 .. +7], p1 supplies K[16+khalf*8 .. +7] per the ISA
// 16-bit A/B VGPR layout.
__device__ __forceinline__ v16bf load_frag(const unsigned short* p0,
                                           const unsigned short* p1) {
  union { u32x4 q[2]; v16bf v; } u;
  u.q[0] = *reinterpret_cast<const u32x4*>(p0);
  u.q[1] = *reinterpret_cast<const u32x4*>(p1);
  return u.v;
}

__device__ __forceinline__ v8f wmma_bf16(v16bf a, v16bf b, v8f c) {
  return __builtin_amdgcn_wmma_f32_16x16x32_bf16(false, a, false, b,
                                                 (short)0, c, false, false);
}

// 16-byte global -> LDS copy: async DMA when available, else direct.
__device__ __forceinline__ void copy16_to_lds(const unsigned short* g,
                                              unsigned short* l) {
#if HAVE_ASYNC_LDS
  __builtin_amdgcn_global_load_async_to_lds_b128(
      (GLOBAL_AS i32x4*)g, (LDS_AS i32x4*)l, 0, 0);
#else
  *reinterpret_cast<u32x4*>(l) = *reinterpret_cast<const u32x4*>(g);
#endif
}

__device__ __forceinline__ void wait_async() {
#if defined(__has_builtin)
#if __has_builtin(__builtin_amdgcn_s_wait_asynccnt)
  __builtin_amdgcn_s_wait_asynccnt(0);
#else
  asm volatile("s_wait_asynccnt 0" ::: "memory");
#endif
#else
  asm volatile("s_wait_asynccnt 0" ::: "memory");
#endif
}

// ---------------------------------------------------------------- converts
__global__ void f2bf_kernel(const float* __restrict__ in,
                            unsigned short* __restrict__ out, int n) {
  int i = blockIdx.x * blockDim.x + threadIdx.x;
  int stride = gridDim.x * blockDim.x;
  for (; i < n; i += stride) out[i] = f2bf(in[i]);
}

// ---------------------------------------------------------------- layernorm
__global__ __launch_bounds__(256) void ln_kernel(
    const float* __restrict__ x, const float* __restrict__ g,
    const float* __restrict__ b, unsigned short* __restrict__ out) {
  __shared__ float red[16];
  const int row = blockIdx.x;
  const float* xr = x + (size_t)row * D_;
  const int base = threadIdx.x * 4;
  float v[4];
#pragma unroll
  for (int i = 0; i < 4; ++i) v[i] = xr[base + i];

  float s = v[0] + v[1] + v[2] + v[3];
#pragma unroll
  for (int off = 16; off > 0; off >>= 1) s += __shfl_xor(s, off);
  const int wid = threadIdx.x >> 5;
  if ((threadIdx.x & 31) == 0) red[wid] = s;
  __syncthreads();
  if (threadIdx.x == 0) {
    float t = 0.f;
    for (int i = 0; i < 8; ++i) t += red[i];
    red[8] = t;
  }
  __syncthreads();
  const float mu = red[8] * (1.0f / D_);

  float sq = 0.f;
#pragma unroll
  for (int i = 0; i < 4; ++i) { float d = v[i] - mu; sq += d * d; }
#pragma unroll
  for (int off = 16; off > 0; off >>= 1) sq += __shfl_xor(sq, off);
  if ((threadIdx.x & 31) == 0) red[wid] = sq;
  __syncthreads();
  if (threadIdx.x == 0) {
    float t = 0.f;
    for (int i = 0; i < 8; ++i) t += red[i];
    red[9] = t;
  }
  __syncthreads();
  const float var = red[9] * (1.0f / D_);
  const float rs = rsqrtf(var + 1e-5f);

  unsigned short* orow = out + (size_t)row * D_;
#pragma unroll
  for (int i = 0; i < 4; ++i)
    orow[base + i] = f2bf((v[i] - mu) * rs * g[base + i] + b[base + i]);
}

// ---------------------------------------------------------------- GEMM
// C[M,N] = A_bf16[M,K] @ W_bf16[N,K]^T + bias, epilogue per MODE:
//   0: bf16 out          (QKV)
//   1: f32 out + resid   (attn projection -> residual2)
//   2: bf16 out, tanh    (fc1)
//   3: f32 out           (fc2 -> final h)
// Block = 256 threads (8 waves); wave -> 16x64 tile (4 accumulators).
// W tile (64 n-rows x 32 k) staged through LDS with double-buffered async DMA.
template <int MODE>
__global__ __launch_bounds__(256) void gemm_kernel(
    const unsigned short* __restrict__ A, const unsigned short* __restrict__ W,
    const float* __restrict__ bias, const float* __restrict__ resid,
    unsigned short* __restrict__ out_bf, float* __restrict__ out_f32,
    int N, int K) {
  __shared__ alignas(16) unsigned short wtile[2][64 * 32];  // 8 KB
  const int tid  = threadIdx.x;
  const int lane = tid & 31;
  const int half = lane >> 4;
  const int lr   = lane & 15;
  const int n0 = blockIdx.x * 64;
  const int m0 = blockIdx.y * 128 + (tid >> 5) * 16;

  // Staging map: 256 threads x 16 B = 4 KB tile. thread -> (row, 16B chunk)
  const int srow   = tid >> 2;
  const int schunk = (tid & 3) * 8;
  const unsigned short* wsrc = W + (size_t)(n0 + srow) * K + schunk;

  v8f acc[4];
#pragma unroll
  for (int t = 0; t < 4; ++t)
#pragma unroll
    for (int r = 0; r < 8; ++r) acc[t][r] = 0.0f;

  const unsigned short* arow = A + (size_t)(m0 + lr) * K;

  const int nk = K >> 5;
  copy16_to_lds(wsrc, &wtile[0][srow * 32 + schunk]);
  for (int i = 0; i < nk; ++i) {
    wait_async();
    __syncthreads();   // tile i visible to all waves; tile i-1 reads retired
    const int k0 = i << 5;
    if (i + 1 < nk)
      copy16_to_lds(wsrc + k0 + 32, &wtile[(i + 1) & 1][srow * 32 + schunk]);

    __builtin_prefetch(arow + k0 + 256, 0, 1);
    v16bf a = load_frag(arow + k0 + half * 8, arow + k0 + 16 + half * 8);
    const unsigned short* wb = wtile[i & 1];
#pragma unroll
    for (int t = 0; t < 4; ++t) {
      const unsigned short* wp = wb + (t * 16 + lr) * 32;
      v16bf b = load_frag(wp + half * 8, wp + 16 + half * 8);
      acc[t] = wmma_bf16(a, b, acc[t]);
    }
  }

#pragma unroll
  for (int t = 0; t < 4; ++t) {
    const int col = n0 + t * 16 + lr;
    const float bv = bias[col];
#pragma unroll
    for (int r = 0; r < 8; ++r) {
      const int row = m0 + r + half * 8;
      const size_t idx = (size_t)row * N + col;
      float v = acc[t][r] + bv;
      if (MODE == 0) out_bf[idx] = f2bf(v);
      else if (MODE == 1) out_f32[idx] = v + resid[idx];
      else if (MODE == 2) out_bf[idx] = f2bf(tanhf(v));
      else out_f32[idx] = v;
    }
  }
}

// ---------------------------------------------------------------- k/v head-sum
// ksum: [m][hd] row-major (B-operand for scores, contiguous along hd)
// vsumT: [b][hd][s] transposed (B-operand for P@V, contiguous along s)
__global__ __launch_bounds__(256) void sumkv_kernel(
    const unsigned short* __restrict__ qkv,
    unsigned short* __restrict__ ksum, unsigned short* __restrict__ vsumT) {
  const int idx = blockIdx.x * blockDim.x + threadIdx.x;  // M_*HD_ threads
  const int m = idx >> 6;
  const int hd = idx & 63;
  const unsigned short* base = qkv + (size_t)m * QKVN_;
  float ks = 0.f, vs = 0.f;
#pragma unroll
  for (int h = 0; h < H_; ++h) {
    ks += bf2f(base[D_ + h * HD_ + hd]);
    vs += bf2f(base[2 * D_ + h * HD_ + hd]);
  }
  const int b = m >> 11;        // m / S_
  const int s = m & (S_ - 1);   // m % S_
  ksum[(size_t)m * HD_ + hd] = f2bf(ks);
  vsumT[(size_t)b * HD_ * S_ + (size_t)hd * S_ + s] = f2bf(vs);
}

// ---------------------------------------------------------------- flash attention
// One wave per (b, h, 16-row q tile). scores = q(16x64) @ ksum^T(64xT) * 0.125,
// causal mask, online softmax, O += P_bf16 @ vsum (N=64).
__global__ __launch_bounds__(32) void attn_kernel(
    const unsigned short* __restrict__ qkv,
    const unsigned short* __restrict__ ksum,
    const unsigned short* __restrict__ vsumT,
    unsigned short* __restrict__ attn_out) {
  __shared__ alignas(16) unsigned short ptile[16 * 32];  // P layout shuffle
  const int lane = threadIdx.x;
  const int half = lane >> 4;
  const int lr   = lane & 15;
  const int QT = S_ / 16;
  const int blk = blockIdx.x;
  const int b = blk / (H_ * QT);
  const int rem = blk % (H_ * QT);
  const int h = rem / QT;
  const int m0 = (rem % QT) * 16;
  const size_t rowbase = (size_t)b * S_;

  // Q fragments (A layout), K = 0..31 and 32..63 of head h
  const unsigned short* qrow = qkv + (rowbase + m0 + lr) * QKVN_ + h * HD_;
  const v16bf qa0 = load_frag(qrow + half * 8, qrow + 16 + half * 8);
  const v16bf qa1 = load_frag(qrow + 32 + half * 8, qrow + 48 + half * 8);

  float mrun[8], lrun[8];
  v8f o[4];
#pragma unroll
  for (int r = 0; r < 8; ++r) { mrun[r] = -3.0e38f; lrun[r] = 0.0f; }
#pragma unroll
  for (int dg = 0; dg < 4; ++dg)
#pragma unroll
    for (int r = 0; r < 8; ++r) o[dg][r] = 0.0f;

  const float scale = 0.125f;  // 1/sqrt(64)
  const int smax = m0 + 15;
  const unsigned short* vtb = vsumT + (size_t)b * HD_ * S_;

  for (int t0 = 0; t0 <= smax; t0 += 32) {
    float sc[2][8];
#pragma unroll
    for (int sub = 0; sub < 2; ++sub) {
      const int tc = t0 + sub * 16;
      if (tc <= smax) {
        v8f s8;
#pragma unroll
        for (int r = 0; r < 8; ++r) s8[r] = 0.0f;
        const unsigned short* krow = ksum + (rowbase + tc + lr) * HD_;
        v16bf kb0 = load_frag(krow + half * 8, krow + 16 + half * 8);
        v16bf kb1 = load_frag(krow + 32 + half * 8, krow + 48 + half * 8);
        s8 = wmma_bf16(qa0, kb0, s8);
        s8 = wmma_bf16(qa1, kb1, s8);
        const int col = tc + lr;
#pragma unroll
        for (int r = 0; r < 8; ++r) {
          const int row = m0 + r + half * 8;
          sc[sub][r] = (col > row) ? -3.0e38f : s8[r] * scale;
        }
      } else {
#pragma unroll
        for (int r = 0; r < 8; ++r) sc[sub][r] = -3.0e38f;
      }
    }

    float p[2][8];
#pragma unroll
    for (int r = 0; r < 8; ++r) {
      float tm = fmaxf(sc[0][r], sc[1][r]);
      tm = fmaxf(tm, __shfl_xor(tm, 1));
      tm = fmaxf(tm, __shfl_xor(tm, 2));
      tm = fmaxf(tm, __shfl_xor(tm, 4));
      tm = fmaxf(tm, __shfl_xor(tm, 8));   // stays inside 16-lane group
      const float nm = fmaxf(mrun[r], tm);
      const float f = __expf(mrun[r] - nm);
      mrun[r] = nm;
      lrun[r] *= f;
#pragma unroll
      for (int dg = 0; dg < 4; ++dg) o[dg][r] *= f;
      const float p0 = __expf(sc[0][r] - nm);
      const float p1 = __expf(sc[1][r] - nm);
      p[0][r] = p0; p[1][r] = p1;
      float ps = p0 + p1;
      ps += __shfl_xor(ps, 1);
      ps += __shfl_xor(ps, 2);
      ps += __shfl_xor(ps, 4);
      ps += __shfl_xor(ps, 8);
      lrun[r] += ps;
    }

    // P: C-layout -> row-major LDS -> A-layout fragment
    __syncthreads();
#pragma unroll
    for (int sub = 0; sub < 2; ++sub)
#pragma unroll
      for (int r = 0; r < 8; ++r)
        ptile[(r + half * 8) * 32 + sub * 16 + lr] = f2bf(p[sub][r]);
    __syncthreads();
    const v16bf pa = load_frag(&ptile[lr * 32 + half * 8],
                               &ptile[lr * 32 + 16 + half * 8]);

#pragma unroll
    for (int dg = 0; dg < 4; ++dg) {
      // B operand column n = hd (dg*16+lr): contiguous along t in vsumT
      const unsigned short* vrow = vtb + (size_t)(dg * 16 + lr) * S_ + t0;
      v16bf vb = load_frag(vrow + half * 8, vrow + 16 + half * 8);
      o[dg] = wmma_bf16(pa, vb, o[dg]);
    }
  }

#pragma unroll
  for (int dg = 0; dg < 4; ++dg)
#pragma unroll
    for (int r = 0; r < 8; ++r) {
      const int row = m0 + r + half * 8;
      const int col = h * HD_ + dg * 16 + lr;
      attn_out[(rowbase + row) * D_ + col] = f2bf(o[dg][r] / lrun[r]);
    }
}

// ---------------------------------------------------------------- launch
extern "C" void kernel_launch(void* const* d_in, const int* in_sizes, int n_in,
                              void* d_out, int out_size, void* d_ws,
                              size_t ws_size, hipStream_t stream) {
  const float* hs   = (const float*)d_in[0];
  const float* Wqkv = (const float*)d_in[1];
  const float* bqkv = (const float*)d_in[2];
  const float* Wout = (const float*)d_in[3];
  const float* bout = (const float*)d_in[4];
  const float* g1   = (const float*)d_in[5];
  const float* b1   = (const float*)d_in[6];
  const float* g2   = (const float*)d_in[7];
  const float* b2   = (const float*)d_in[8];
  const float* W1   = (const float*)d_in[9];
  const float* bfc1 = (const float*)d_in[10];
  const float* W2   = (const float*)d_in[11];
  const float* bfc2 = (const float*)d_in[12];

  float* out_h     = (float*)d_out;                    // (B,S,D) mlp output
  float* out_resid = out_h + (size_t)M_ * D_;          // (B,S,D) residual

  char* ws = (char*)d_ws;
  size_t off = 0;
  auto alloc_u16 = [&](size_t elems) -> unsigned short* {
    unsigned short* p = (unsigned short*)(ws + off);
    off += ((elems * 2 + 255) / 256) * 256;
    return p;
  };
  unsigned short* ln1   = alloc_u16((size_t)M_ * D_);
  unsigned short* wqkvb = alloc_u16((size_t)QKVN_ * D_);
  unsigned short* woutb = alloc_u16((size_t)D_ * D_);
  unsigned short* w1b   = alloc_u16((size_t)4 * D_ * D_);
  unsigned short* w2b   = alloc_u16((size_t)4 * D_ * D_);
  unsigned short* qkv   = alloc_u16((size_t)M_ * QKVN_);
  unsigned short* ksum  = alloc_u16((size_t)M_ * HD_);
  unsigned short* vsumT = alloc_u16((size_t)M_ * HD_);
  unsigned short* attn  = alloc_u16((size_t)M_ * D_);
  unsigned short* ln2   = alloc_u16((size_t)M_ * D_);
  unsigned short* h1    = alloc_u16((size_t)M_ * 4 * D_);
  (void)ws_size; (void)in_sizes; (void)n_in; (void)out_size;

  // Weight fp32 -> bf16
  f2bf_kernel<<<2048, 256, 0, stream>>>(Wqkv, wqkvb, QKVN_ * D_);
  f2bf_kernel<<<2048, 256, 0, stream>>>(Wout, woutb, D_ * D_);
  f2bf_kernel<<<2048, 256, 0, stream>>>(W1, w1b, 4 * D_ * D_);
  f2bf_kernel<<<2048, 256, 0, stream>>>(W2, w2b, 4 * D_ * D_);

  // LN1
  ln_kernel<<<M_, 256, 0, stream>>>(hs, g1, b1, ln1);

  // QKV projection (+bias) -> bf16
  gemm_kernel<0><<<dim3(QKVN_ / 64, M_ / 128), 256, 0, stream>>>(
      ln1, wqkvb, bqkv, nullptr, qkv, nullptr, QKVN_, D_);

  // k_sum / v_sum over heads
  sumkv_kernel<<<(M_ * HD_) / 256, 256, 0, stream>>>(qkv, ksum, vsumT);

  // Flash attention
  attn_kernel<<<B_ * H_ * (S_ / 16), 32, 0, stream>>>(qkv, ksum, vsumT, attn);

  // Output projection + bias + residual -> fp32 residual2 (also final output)
  gemm_kernel<1><<<dim3(D_ / 64, M_ / 128), 256, 0, stream>>>(
      attn, woutb, bout, hs, nullptr, out_resid, D_, D_);

  // LN2 on residual2
  ln_kernel<<<M_, 256, 0, stream>>>(out_resid, g2, b2, ln2);

  // FC1 (+bias, tanh) -> bf16
  gemm_kernel<2><<<dim3(4 * D_ / 64, M_ / 128), 256, 0, stream>>>(
      ln2, w1b, bfc1, nullptr, h1, nullptr, 4 * D_, D_);

  // FC2 (+bias) -> fp32 h output
  gemm_kernel<3><<<dim3(D_ / 64, M_ / 128), 256, 0, stream>>>(
      h1, w2b, bfc2, nullptr, nullptr, out_h, D_, 4 * D_);
}
